// multihead_self_attention_10625749090406
// MI455X (gfx1250) — matrix operand order repaired and checked by reading, not compiled, not run on hardware
//
#include <hip/hip_runtime.h>

#define D_MODELc 2048
#define NUM_HEADSc 16
#define D_HEADc 128
#define Bc 2
#define Tc 2048

typedef __attribute__((ext_vector_type(8)))  __bf16 v8bf;
typedef __attribute__((ext_vector_type(16))) __bf16 v16bf;
typedef __attribute__((ext_vector_type(8)))  float  v8f;

static __device__ __forceinline__ v8bf ldg8(const __bf16* p) {
    return *reinterpret_cast<const v8bf*>(p);
}

static __device__ __forceinline__ v16bf cat16(v8bf lo, v8bf hi) {
    v16bf r;
#pragma unroll
    for (int i = 0; i < 8; ++i) { r[i] = lo[i]; r[i + 8] = hi[i]; }
    return r;
}

static __device__ __forceinline__ v8f wmma_bf16(v16bf a, v16bf b, v8f c) {
    // D = A(16x32 bf16) x B(32x16 bf16) + C(16x16 f32)
    return __builtin_amdgcn_wmma_f32_16x16x32_bf16(
        /*neg_a=*/false, a, /*neg_b=*/false, b,
        /*c_mod=*/(short)0, c, /*reuse_a=*/false, /*reuse_b=*/false);
}

// Issue one 16B/lane asynchronous copy global -> LDS (ASYNCcnt-tracked).
static __device__ __forceinline__ void async_ld16(unsigned lds_off,
                                                  const __bf16* gptr) {
    asm volatile("global_load_async_to_lds_b128 %0, %1, off"
                 :: "v"(lds_off), "v"(gptr) : "memory");
}

static __device__ __forceinline__ unsigned lds_offset(const void* p) {
    // generic LDS pointer: addr[31:0] is the LDS byte offset (ISA 10.2)
    return (unsigned)(uintptr_t)p;
}

// ---------------------------------------------------------------------------
// Kernel 1: fp32 -> bf16 conversion (grid-stride)
// ---------------------------------------------------------------------------
__global__ void cvt_f32_bf16(const float* __restrict__ src,
                             __bf16* __restrict__ dst, size_t n) {
    for (size_t i = (size_t)blockIdx.x * blockDim.x + threadIdx.x; i < n;
         i += (size_t)gridDim.x * blockDim.x) {
        dst[i] = (__bf16)src[i];
    }
}

// ---------------------------------------------------------------------------
// Kernel 2: QKV projection GEMM.  out = X(4096x2048) @ W(2048x2048)^T
// Block = 4 waves, block tile 128x64. Wave tile 32x64 (8 accumulators).
// W tile staged via async global->LDS (double buffered, k-loop manually
// unrolled x2: ping-pong without register copies). All 4 B-fragments are
// loaded from LDS as one clause, then 8 WMMAs issue back-to-back.
// vmode==0 : out[(b*H+h)*T + t, d]      (Q,K layout [B,H,T,Dh])
// vmode==1 : out[(b*H+h)*Dh + d, t]     (V transposed  [B,H,Dh,T])
// ---------------------------------------------------------------------------
__global__ __launch_bounds__(128) void gemm_qkv(const __bf16* __restrict__ X,
                                                const __bf16* __restrict__ W,
                                                __bf16* __restrict__ out,
                                                int vmode) {
    __shared__ __bf16 sB[2][64][40];  // padded rows: stride 20 dwords

    const int td = threadIdx.x;
    const int lane = td & 31;
    const int wave = td >> 5;
    const int lm = lane & 15;
    const int hs = lane >> 4;
    const int Mbase = blockIdx.x * 128 + wave * 32;
    const int Nbase = blockIdx.y * 64;
    const int row0 = Mbase + lm;       // A-frag rows: M = lane&15
    const int row1 = Mbase + 16 + lm;

    auto fill = [&](int buf, int k) {  // 2 async ops per wave, 4KB per block
#pragma unroll
        for (int j = 0; j < 2; ++j) {
            const int c = j * 128 + td;        // 256 x 16B chunks
            const int r = c >> 2, c8 = c & 3;  // row 0..63, 8-col group
            async_ld16(lds_offset(&sB[buf][r][c8 * 8]),
                       &W[(size_t)(Nbase + r) * D_MODELc + k + c8 * 8]);
        }
    };
    auto loadA = [&](int row, int k) {
        return cat16(ldg8(&X[(size_t)row * D_MODELc + k + hs * 16]),
                     ldg8(&X[(size_t)row * D_MODELc + k + hs * 16 + 8]));
    };

    v8f acc[2][4] = {};
    fill(0, 0);
    v16bf a0 = loadA(row0, 0);
    v16bf a1 = loadA(row1, 0);
    v16bf n0 = a0, n1 = a1;

    // compute one k-step from LDS buffer `buf` with fragments x0/x1
    auto compute = [&](int buf, const v16bf& x0, const v16bf& x1) {
        v16bf bfr[4];
#pragma unroll
        for (int n = 0; n < 4; ++n)  // 8 ds_load_b128 as one clause
            bfr[n] = cat16(*(const v8bf*)&sB[buf][n * 16 + lm][hs * 16],
                           *(const v8bf*)&sB[buf][n * 16 + lm][hs * 16 + 8]);
#pragma unroll
        for (int n = 0; n < 4; ++n) {  // 8 back-to-back WMMAs
            acc[0][n] = wmma_bf16(x0, bfr[n], acc[0][n]);
            acc[1][n] = wmma_bf16(x1, bfr[n], acc[1][n]);
        }
    };

    constexpr int STEPS = D_MODELc / 32;  // 64, even
#pragma unroll 1
    for (int kk = 0; kk < STEPS; kk += 2) {
        // ---- even step kk: buffer 0, fragments a0/a1 ----------------------
        {
            const int kn = (kk + 1) * 32;  // kk+1 < STEPS always
            fill(1, kn);
            __builtin_prefetch(&X[(size_t)row0 * D_MODELc + kn + 32], 0, 0);
            n0 = loadA(row0, kn);
            n1 = loadA(row1, kn);
            asm volatile("s_wait_asynccnt 0x2" ::: "memory");
        }
        __syncthreads();
        compute(0, a0, a1);
        __syncthreads();
        // ---- odd step kk+1: buffer 1, fragments n0/n1 ---------------------
        if (kk + 2 < STEPS) {
            const int kn = (kk + 2) * 32;
            fill(0, kn);
            a0 = loadA(row0, kn);
            a1 = loadA(row1, kn);
            asm volatile("s_wait_asynccnt 0x2" ::: "memory");
        } else {
            asm volatile("s_wait_asynccnt 0x0" ::: "memory");
        }
        __syncthreads();
        compute(1, n0, n1);
        __syncthreads();
    }

#pragma unroll
    for (int m = 0; m < 2; ++m) {
#pragma unroll
        for (int n = 0; n < 4; ++n) {
#pragma unroll
            for (int r = 0; r < 8; ++r) {
                const int row = Mbase + m * 16 + r + hs * 8;  // C/D: M=r+8*hs
                const int bb = row >> 11;
                const int t = row & (Tc - 1);
                const int col = Nbase + n * 16 + lm;          // C/D: N=lane&15
                const int h = col >> 7;
                const int d = col & (D_HEADc - 1);
                const __bf16 v = (__bf16)acc[m][n][r];
                if (vmode)
                    out[((size_t)(bb * NUM_HEADSc + h) * D_HEADc + d) * Tc + t] = v;
                else
                    out[((size_t)(bb * NUM_HEADSc + h) * Tc + t) * D_HEADc + d] = v;
            }
        }
    }
}

// ---------------------------------------------------------------------------
// Kernel 3: RoPE in place on Q and K, layout [B,H,T,Dh] (pairs contiguous)
// ---------------------------------------------------------------------------
__global__ void rope_kernel(__bf16* __restrict__ Q, __bf16* __restrict__ K,
                            const int* __restrict__ pos) {
    const size_t total = (size_t)Bc * NUM_HEADSc * Tc * (D_HEADc / 2);
    size_t idx = (size_t)blockIdx.x * blockDim.x + threadIdx.x;
    if (idx >= total) return;
    const int i = (int)(idx & (D_HEADc / 2 - 1));  // pair index 0..63
    const size_t bht = idx >> 6;
    const int t = (int)(bht & (Tc - 1));
    const size_t base = bht * D_HEADc;

    const float p = (float)pos[t];
    // inv_freq = theta^(-2i/Dh) ; log(10000) = 9.210340372
    const float ang = p * __expf(-(2.0f * (float)i / (float)D_HEADc) * 9.210340372f);
    const float c = __cosf(ang);
    const float s = __sinf(ang);

    const float q1 = (float)Q[base + 2 * i], q2 = (float)Q[base + 2 * i + 1];
    Q[base + 2 * i]     = (__bf16)(q1 * c - q2 * s);
    Q[base + 2 * i + 1] = (__bf16)(q1 * s + q2 * c);
    const float k1 = (float)K[base + 2 * i], k2 = (float)K[base + 2 * i + 1];
    K[base + 2 * i]     = (__bf16)(k1 * c - k2 * s);
    K[base + 2 * i + 1] = (__bf16)(k1 * s + k2 * c);
}

// ---------------------------------------------------------------------------
// Kernel 4: causal flash attention. grid = (B*H, T/64), block = 128 (4 waves).
// Each wave owns 16 query rows; K/V key-chunks (32 keys) are staged into LDS
// with async copies (double buffered, shared by all 4 waves). Q fragments are
// loaded once (invariant over the chunk loop). Online softmax per wave.
// ---------------------------------------------------------------------------
__global__ __launch_bounds__(128) void attn_kernel(const __bf16* __restrict__ Q,
                                                   const __bf16* __restrict__ K,
                                                   const __bf16* __restrict__ Vt,
                                                   __bf16* __restrict__ O) {
    __shared__ __bf16 sK[2][32][136];  // 32 keys x 128 dims, rows padded (+8)
    __shared__ __bf16 sV[2][128][40];  // 128 dims x 32 keys, rows padded (+8)
    __shared__ float sS[4][16][32];
    __shared__ float sP[4][16][32];
    __shared__ float sA[4][16];
    __shared__ float sL[4][16];

    const int td = threadIdx.x;
    const int lane = td & 31;
    const int wave = td >> 5;
    const int lm = lane & 15;
    const int hs = lane >> 4;
    const int bh = blockIdx.x;
    const int b = bh >> 4;
    const int h = bh & (NUM_HEADSc - 1);
    const int qBlock = blockIdx.y * 64;
    const int qBase = qBlock + wave * 16;
    const float scale = 0.08838834764831845f;  // 1/sqrt(128)

    const size_t qk_base = (size_t)bh * Tc * D_HEADc;
    const size_t vt_base = (size_t)bh * D_HEADc * Tc;

    auto fillKV = [&](int buf, int keyBase) {  // 8 async ops per wave (16KB)
#pragma unroll
        for (int j = 0; j < 4; ++j) {          // K tile: 32 rows x 128 dims
            const int c = j * 128 + td;
            const int r = c >> 4, c16 = c & 15;
            async_ld16(lds_offset(&sK[buf][r][c16 * 8]),
                       &K[qk_base + (size_t)(keyBase + r) * D_HEADc + c16 * 8]);
        }
#pragma unroll
        for (int j = 0; j < 4; ++j) {          // V tile: 128 rows x 32 keys
            const int c = j * 128 + td;
            const int r = c >> 2, c4 = c & 3;
            async_ld16(lds_offset(&sV[buf][r][c4 * 8]),
                       &Vt[vt_base + (size_t)r * Tc + keyBase + c4 * 8]);
        }
    };

    // Q fragments: invariant across the key loop, load once.
    v16bf qa[4];
#pragma unroll
    for (int dk = 0; dk < 4; ++dk) {
        const int d0 = dk * 32;
        qa[dk] = cat16(
            ldg8(&Q[qk_base + (size_t)(qBase + lm) * D_HEADc + d0 + hs * 16]),
            ldg8(&Q[qk_base + (size_t)(qBase + lm) * D_HEADc + d0 + hs * 16 + 8]));
    }

    v8f acc[8] = {};
    float m_i = -INFINITY, l_i = 0.0f;  // live in lanes 0..15, row = lane

    const int nChunks = (qBlock + 64) / 32;  // causal bound, block-uniform
    fillKV(0, 0);
#pragma unroll 1
    for (int c = 0; c < nChunks; ++c) {
        const int keyBase = c * 32;
        const int cur = c & 1;
        if (c + 1 < nChunks) {
            fillKV(1 - cur, keyBase + 32);  // overlap next fill with compute
            asm volatile("s_wait_asynccnt 0x8" ::: "memory");
        } else {
            asm volatile("s_wait_asynccnt 0x0" ::: "memory");
        }
        __syncthreads();  // cur K/V tiles visible to all waves
        // ---- S = Q K^T over Dh=128: 2 groups x (8 ds loads + 4 WMMAs) -----
        v8f s0 = {}, s1 = {};
#pragma unroll
        for (int g = 0; g < 2; ++g) {
            v16bf kb[4];
#pragma unroll
            for (int d = 0; d < 2; ++d) {
                const int d0 = (g * 2 + d) * 32;
                kb[d * 2] = cat16(*(const v8bf*)&sK[cur][lm][d0 + hs * 16],
                                  *(const v8bf*)&sK[cur][lm][d0 + hs * 16 + 8]);
                kb[d * 2 + 1] =
                    cat16(*(const v8bf*)&sK[cur][16 + lm][d0 + hs * 16],
                          *(const v8bf*)&sK[cur][16 + lm][d0 + hs * 16 + 8]);
            }
#pragma unroll
            for (int d = 0; d < 2; ++d) {
                s0 = wmma_bf16(qa[g * 2 + d], kb[d * 2], s0);
                s1 = wmma_bf16(qa[g * 2 + d], kb[d * 2 + 1], s1);
            }
        }
        // ---- spill S tile to this wave's LDS slice ------------------------
#pragma unroll
        for (int r = 0; r < 8; ++r) {
            const int row = r + hs * 8;
            sS[wave][row][lm]      = s0[r] * scale;
            sS[wave][row][lm + 16] = s1[r] * scale;
        }
        asm volatile("s_wait_dscnt 0x0" ::: "memory");
        __builtin_amdgcn_wave_barrier();
        // ---- online softmax: lanes 0..15, one query row each --------------
        if (lane < 16) {
            const int qg = qBase + lane;
            float rmax = -INFINITY;
#pragma unroll
            for (int j = 0; j < 32; ++j) {
                const float v = (keyBase + j <= qg) ? sS[wave][lane][j] : -INFINITY;
                rmax = fmaxf(rmax, v);
            }
            const float mnew = fmaxf(m_i, rmax);
            const float alpha = __expf(m_i - mnew);
            float lsum = 0.0f;
#pragma unroll
            for (int j = 0; j < 32; ++j) {
                const float v = (keyBase + j <= qg) ? sS[wave][lane][j] : -INFINITY;
                const float p = __expf(v - mnew);
                sP[wave][lane][j] = p;
                lsum += p;
            }
            l_i = l_i * alpha + lsum;
            m_i = mnew;
            sA[wave][lane] = alpha;
        }
        asm volatile("s_wait_dscnt 0x0" ::: "memory");
        __builtin_amdgcn_wave_barrier();
        // ---- rescale output accumulators by alpha[row] --------------------
#pragma unroll
        for (int r = 0; r < 8; ++r) {
            const float alpha = sA[wave][r + hs * 8];
#pragma unroll
            for (int n = 0; n < 8; ++n) acc[n][r] *= alpha;
        }
        // ---- P fragment (A-matrix layout) from LDS, bf16 ------------------
        v16bf pa;
#pragma unroll
        for (int e = 0; e < 16; ++e) {
            const int kk = hs * 16 + e;
            pa[e] = (__bf16)sP[wave][lm][kk];
        }
        // ---- O += P @ V: 2 groups x (8 ds loads + 4 WMMAs) ----------------
#pragma unroll
        for (int g = 0; g < 2; ++g) {
            v16bf bv[4];
#pragma unroll
            for (int n = 0; n < 4; ++n) {
                const int nr = (g * 4 + n) * 16 + lm;
                bv[n] = cat16(*(const v8bf*)&sV[cur][nr][hs * 16],
                              *(const v8bf*)&sV[cur][nr][hs * 16 + 8]);
            }
#pragma unroll
            for (int n = 0; n < 4; ++n)
                acc[g * 4 + n] = wmma_bf16(pa, bv[n], acc[g * 4 + n]);
        }
        __syncthreads();  // all waves done with cur before it is refilled
    }
    // ---- normalize by l and emit [B,T,D_MODEL] bf16 -----------------------
    if (lane < 16) sL[wave][lane] = l_i;
    asm volatile("s_wait_dscnt 0x0" ::: "memory");
    __builtin_amdgcn_wave_barrier();
#pragma unroll
    for (int r = 0; r < 8; ++r) {
        const int row = r + hs * 8;
        const float inv = 1.0f / sL[wave][row];
        const int t = qBase + row;
#pragma unroll
        for (int n = 0; n < 8; ++n) {
            O[(size_t)(b * Tc + t) * D_MODELc + h * D_HEADc + n * 16 + lm] =
                (__bf16)(acc[n][r] * inv);
        }
    }
}

// ---------------------------------------------------------------------------
// Kernel 5: output projection. out(fp32) = O(4096x2048 bf16) @ Wp^T
// Same async-LDS double-buffered, x2-unrolled structure as gemm_qkv.
// ---------------------------------------------------------------------------
__global__ __launch_bounds__(128) void gemm_proj(const __bf16* __restrict__ X,
                                                 const __bf16* __restrict__ W,
                                                 float* __restrict__ out) {
    __shared__ __bf16 sB[2][64][40];

    const int td = threadIdx.x;
    const int lane = td & 31;
    const int wave = td >> 5;
    const int lm = lane & 15;
    const int hs = lane >> 4;
    const int Mbase = blockIdx.x * 128 + wave * 32;
    const int Nbase = blockIdx.y * 64;
    const int row0 = Mbase + lm, row1 = Mbase + 16 + lm;

    auto fill = [&](int buf, int k) {
#pragma unroll
        for (int j = 0; j < 2; ++j) {
            const int c = j * 128 + td;
            const int r = c >> 2, c8 = c & 3;
            async_ld16(lds_offset(&sB[buf][r][c8 * 8]),
                       &W[(size_t)(Nbase + r) * D_MODELc + k + c8 * 8]);
        }
    };
    auto loadA = [&](int row, int k) {
        return cat16(ldg8(&X[(size_t)row * D_MODELc + k + hs * 16]),
                     ldg8(&X[(size_t)row * D_MODELc + k + hs * 16 + 8]));
    };

    v8f acc[2][4] = {};
    fill(0, 0);
    v16bf a0 = loadA(row0, 0);
    v16bf a1 = loadA(row1, 0);
    v16bf n0 = a0, n1 = a1;

    auto compute = [&](int buf, const v16bf& x0, const v16bf& x1) {
        v16bf bfr[4];
#pragma unroll
        for (int n = 0; n < 4; ++n)
            bfr[n] = cat16(*(const v8bf*)&sB[buf][n * 16 + lm][hs * 16],
                           *(const v8bf*)&sB[buf][n * 16 + lm][hs * 16 + 8]);
#pragma unroll
        for (int n = 0; n < 4; ++n) {
            acc[0][n] = wmma_bf16(x0, bfr[n], acc[0][n]);
            acc[1][n] = wmma_bf16(x1, bfr[n], acc[1][n]);
        }
    };

    constexpr int STEPS = D_MODELc / 32;
#pragma unroll 1
    for (int kk = 0; kk < STEPS; kk += 2) {
        {
            const int kn = (kk + 1) * 32;
            fill(1, kn);
            __builtin_prefetch(&X[(size_t)row0 * D_MODELc + kn + 32], 0, 0);
            n0 = loadA(row0, kn);
            n1 = loadA(row1, kn);
            asm volatile("s_wait_asynccnt 0x2" ::: "memory");
        }
        __syncthreads();
        compute(0, a0, a1);
        __syncthreads();
        if (kk + 2 < STEPS) {
            const int kn = (kk + 2) * 32;
            fill(0, kn);
            a0 = loadA(row0, kn);
            a1 = loadA(row1, kn);
            asm volatile("s_wait_asynccnt 0x2" ::: "memory");
        } else {
            asm volatile("s_wait_asynccnt 0x0" ::: "memory");
        }
        __syncthreads();
        compute(1, n0, n1);
        __syncthreads();
    }

#pragma unroll
    for (int m = 0; m < 2; ++m) {
#pragma unroll
        for (int n = 0; n < 4; ++n) {
#pragma unroll
            for (int r = 0; r < 8; ++r) {
                const int row = Mbase + m * 16 + r + hs * 8;
                const int col = Nbase + n * 16 + lm;
                out[(size_t)row * D_MODELc + col] = acc[m][n][r];
            }
        }
    }
}

// ---------------------------------------------------------------------------
extern "C" void kernel_launch(void* const* d_in, const int* in_sizes, int n_in,
                              void* d_out, int out_size, void* d_ws, size_t ws_size,
                              hipStream_t stream) {
    const float* x   = (const float*)d_in[0];
    const int*   pos = (const int*)d_in[1];
    const float* Wq  = (const float*)d_in[2];
    const float* Wk  = (const float*)d_in[3];
    const float* Wv  = (const float*)d_in[4];
    const float* Wp  = (const float*)d_in[5];
    float* out = (float*)d_out;

    const size_t szX = (size_t)Bc * Tc * D_MODELc;    // 8,388,608 elements
    const size_t szW = (size_t)D_MODELc * D_MODELc;   // 4,194,304 elements

    __bf16* xb  = (__bf16*)d_ws;
    __bf16* wqb = xb + szX;
    __bf16* wkb = wqb + szW;
    __bf16* wvb = wkb + szW;
    __bf16* wpb = wvb + szW;
    __bf16* Qb  = wpb + szW;   // [B,H,T,Dh]
    __bf16* Kb  = Qb + szX;    // [B,H,T,Dh]
    __bf16* Vtb = Kb + szX;    // [B,H,Dh,T]
    __bf16* Ob  = Vtb + szX;   // [B,T,D]

    cvt_f32_bf16<<<2048, 256, 0, stream>>>(x,  xb,  szX);
    cvt_f32_bf16<<<1024, 256, 0, stream>>>(Wq, wqb, szW);
    cvt_f32_bf16<<<1024, 256, 0, stream>>>(Wk, wkb, szW);
    cvt_f32_bf16<<<1024, 256, 0, stream>>>(Wv, wvb, szW);
    cvt_f32_bf16<<<1024, 256, 0, stream>>>(Wp, wpb, szW);

    dim3 g(32, 32);  // M tiles (4096/128), N tiles (2048/64)
    gemm_qkv<<<g, 128, 0, stream>>>(xb, wqb, Qb, 0);
    gemm_qkv<<<g, 128, 0, stream>>>(xb, wkb, Kb, 0);
    gemm_qkv<<<g, 128, 0, stream>>>(xb, wvb, Vtb, 1);

    const int pairs = Bc * NUM_HEADSc * Tc * (D_HEADc / 2);
    rope_kernel<<<(pairs + 255) / 256, 256, 0, stream>>>(Qb, Kb, pos);

    attn_kernel<<<dim3(Bc * NUM_HEADSc, Tc / 64), 128, 0, stream>>>(Qb, Kb, Vtb, Ob);

    gemm_proj<<<g, 128, 0, stream>>>(Ob, wpb, out);
}
